// roi_pool_51694226375164
// MI455X (gfx1250) — matrix operand order repaired and checked by reading, not compiled, not run on hardware
//
#include <hip/hip_runtime.h>

typedef __attribute__((ext_vector_type(16))) _Float16 v16h;
typedef __attribute__((ext_vector_type(8)))  float    v8f;

#define NB_      32                      // batches
#define NVOX     50000                   // voxels (K of the GEMM)
#define CIN      64                      // channels (N of the GEMM)
#define NCLUST   50                      // clusters (M of the GEMM, padded to 64)
#define KC       128                     // K-chunk staged per workgroup iteration
#define MROWS    64                      // padded cluster rows
#define XT_STRIDE (KC + 4)               // 132 halves: conflict-free, 8B-aligned rows
#define NCHUNK   ((NVOX + KC - 1) / KC)  // 391
#define NBLK     ((NCHUNK + 3) / 4)      // 98 workgroups, ~4 chunks each
#define PARTIAL_STRIDE (MROWS * CIN + MROWS)  // 4160 floats per block partial

union SMemU {
    struct { _Float16 Xt[CIN * XT_STRIDE];   // x chunk, channel-major f16
             _Float16 Mh[MROWS * XT_STRIDE]; // mask chunk, cluster-major f16
    } s;
    struct { float b0[MROWS * CIN];          // cross-wave reduction buffers
             float b1[MROWS * CIN];
    } r;
};

__global__ void __launch_bounds__(128)
cluster_mean_reduce(const float* __restrict__ x,        // uses only batch 0: [NVOX][CIN]
                    const int*   __restrict__ clusters, // [NCLUST][NVOX]
                    float*       __restrict__ gpart)    // [NBLK][PARTIAL_STRIDE]
{
    __shared__ __align__(16) SMemU smem;

    const int tid  = threadIdx.x;
    const int lane = tid & 31;
    const int wave = tid >> 5;
    const int r16  = lane & 15;
    const int hi   = lane >> 4;

    // 4x4 tile grid of 16x16 f32 accumulators (full padded 64x64 output per wave)
    v8f acc[4][4];
#pragma unroll
    for (int m = 0; m < 4; ++m)
#pragma unroll
        for (int n = 0; n < 4; ++n)
#pragma unroll
            for (int e = 0; e < 8; ++e) acc[m][n][e] = 0.0f;

    float ccount = 0.0f;  // per-cluster-row count accumulator (tid < 64 owns row tid)

    for (int chunk = blockIdx.x; chunk < NCHUNK; chunk += gridDim.x) {
        const int vbase = chunk * KC;

        // Prefetch next chunk this block will process (global_prefetch_b8 path).
        const int vnext = vbase + (int)gridDim.x * KC;
        if (vnext < NVOX) {
            const char* xb = (const char*)(x + vnext * CIN);
            __builtin_prefetch(xb + tid * 256, 0, 1);
            __builtin_prefetch(xb + tid * 256 + 128, 0, 1);
            for (int t = tid; t < NCLUST * 4; t += 128) {
                const char* mb = (const char*)(clusters + (t >> 2) * NVOX + vnext);
                __builtin_prefetch(mb + (t & 3) * 128, 0, 1);
            }
        }

        // Stage x[0] chunk: coalesced global read (lane -> channel),
        // transposed f16 write to channel-major LDS (stride 132 -> no bank conflict).
        for (int idx = tid; idx < KC * CIN; idx += 128) {
            const int v  = idx >> 6;
            const int c  = idx & 63;
            const int gv = vbase + v;
            const float val = (gv < NVOX) ? x[gv * CIN + c] : 0.0f;
            smem.s.Xt[c * XT_STRIDE + v] = (_Float16)val;
        }
        // Stage mask chunk: coalesced along voxels, rows >= NCLUST padded with zeros.
        for (int idx = tid; idx < MROWS * KC; idx += 128) {
            const int v  = idx & (KC - 1);
            const int k  = idx >> 7;
            const int gv = vbase + v;
            float mval = 0.0f;
            if (k < NCLUST && gv < NVOX)
                mval = (clusters[k * NVOX + gv] == 1) ? 1.0f : 0.0f;
            smem.s.Mh[k * XT_STRIDE + v] = (_Float16)mval;
        }
        __syncthreads();

        // Per-cluster counts: free row-sum of the staged mask.
        if (tid < MROWS) {
            const _Float16* row = &smem.s.Mh[tid * XT_STRIDE];
            float s = 0.0f;
            for (int j = 0; j < KC; ++j) s += (float)row[j];
            ccount += s;
        }

        const int kb = wave * 32;  // this wave's K sub-slice within the chunk

        // B fragments (32x16 f16): lane holds column N=r16; halves 0..15 are
        // 16 consecutive K values (lanes 16-31 take K+16). 8B-aligned LDS loads.
        v16h bfrag[4];
#pragma unroll
        for (int n = 0; n < 4; ++n) {
            const _Float16* p = &smem.s.Xt[(16 * n + r16) * XT_STRIDE + kb + hi * 16];
            union { v16h v; uint2 q[4]; } ub;
            ub.q[0] = *(const uint2*)(p + 0);
            ub.q[1] = *(const uint2*)(p + 4);
            ub.q[2] = *(const uint2*)(p + 8);
            ub.q[3] = *(const uint2*)(p + 12);
            bfrag[n] = ub.v;
        }

        // A fragments (16x32 f16): lane holds row M=r16; halves 0..15 are
        // 16 consecutive K values kb+hi*16.. (same K order as B).
#pragma unroll
        for (int m = 0; m < 4; ++m) {
            const _Float16* p = &smem.s.Mh[(16 * m + r16) * XT_STRIDE + kb + hi * 16];
            union { v16h v; uint2 q[4]; } ua;
            ua.q[0] = *(const uint2*)(p + 0);
            ua.q[1] = *(const uint2*)(p + 4);
            ua.q[2] = *(const uint2*)(p + 8);
            ua.q[3] = *(const uint2*)(p + 12);
#pragma unroll
            for (int n = 0; n < 4; ++n) {
                acc[m][n] = __builtin_amdgcn_wmma_f32_16x16x32_f16(
                    false, ua.v, false, bfrag[n], (short)0, acc[m][n], false, false);
            }
        }
        __syncthreads();  // protect LDS before next chunk's staging
    }

    // Deterministic cross-wave reduction (no float atomics -> replay-stable).
    // C/D layout: VGPR e -> row 16m + hi*8 + e, col 16n + r16.
    if (wave == 0) {
#pragma unroll
        for (int m = 0; m < 4; ++m)
#pragma unroll
            for (int n = 0; n < 4; ++n)
#pragma unroll
                for (int e = 0; e < 8; ++e)
                    smem.r.b0[(16 * m + hi * 8 + e) * CIN + 16 * n + r16] = acc[m][n][e];
    } else if (wave == 1) {
#pragma unroll
        for (int m = 0; m < 4; ++m)
#pragma unroll
            for (int n = 0; n < 4; ++n)
#pragma unroll
                for (int e = 0; e < 8; ++e)
                    smem.r.b1[(16 * m + hi * 8 + e) * CIN + 16 * n + r16] = acc[m][n][e];
    }
    __syncthreads();
    if (wave == 2) {
#pragma unroll
        for (int m = 0; m < 4; ++m)
#pragma unroll
            for (int n = 0; n < 4; ++n)
#pragma unroll
                for (int e = 0; e < 8; ++e)
                    smem.r.b0[(16 * m + hi * 8 + e) * CIN + 16 * n + r16] += acc[m][n][e];
    } else if (wave == 3) {
#pragma unroll
        for (int m = 0; m < 4; ++m)
#pragma unroll
            for (int n = 0; n < 4; ++n)
#pragma unroll
                for (int e = 0; e < 8; ++e)
                    smem.r.b1[(16 * m + hi * 8 + e) * CIN + 16 * n + r16] += acc[m][n][e];
    }
    __syncthreads();

    float* partial = gpart + (size_t)blockIdx.x * PARTIAL_STRIDE;
    for (int idx = tid; idx < MROWS * CIN; idx += 128)
        partial[idx] = smem.r.b0[idx] + smem.r.b1[idx];
    if (tid < MROWS)
        partial[MROWS * CIN + tid] = ccount;
}

// Sum the NBLK partials deterministically, divide by counts, broadcast over batches.
__global__ void __launch_bounds__(256)
cluster_mean_finalize(const float* __restrict__ gpart, float* __restrict__ out)
{
    const int i = blockIdx.x * blockDim.x + threadIdx.x;  // 0 .. NCLUST*CIN-1
    if (i >= NCLUST * CIN) return;
    const int k = i / CIN;
    const int c = i % CIN;
    float s = 0.0f, cnt = 0.0f;
    for (int b = 0; b < NBLK; ++b) {
        const float* p = gpart + (size_t)b * PARTIAL_STRIDE;
        s   += p[k * CIN + c];
        cnt += p[MROWS * CIN + k];
    }
    const float mean = s / cnt;
#pragma unroll
    for (int bb = 0; bb < NB_; ++bb)
        out[bb * NCLUST * CIN + i] = mean;
}

extern "C" void kernel_launch(void* const* d_in, const int* in_sizes, int n_in,
                              void* d_out, int out_size, void* d_ws, size_t ws_size,
                              hipStream_t stream) {
    const float* x        = (const float*)d_in[0];   // [32, 50000, 64] f32 (batch 0 used)
    const int*   clusters = (const int*)d_in[1];     // [50, 50000] i32
    float*       out      = (float*)d_out;           // [32, 50, 64] f32
    float*       gpart    = (float*)d_ws;            // NBLK * 4160 floats (~1.6 MB)

    cluster_mean_reduce<<<NBLK, 128, 0, stream>>>(x, clusters, gpart);
    cluster_mean_finalize<<<(NCLUST * CIN + 255) / 256, 256, 0, stream>>>(gpart, out);
}
